// CrossAttention_14663018348623
// MI455X (gfx1250) — compile-verified, matrix operand order checked by reading, not run on hardware
//
#include <hip/hip_runtime.h>

// ---------------------------------------------------------------------------
// CDNA5 (gfx1250, wave32) cross-attention:
//   Q=S2*Wq, K=S1*Wk, V=S1*Wv, S=Q*K^T, P=softmax(S), M=P*V, out=M*Wo+bo
// All GEMMs: LDS-staged, double-buffered v_wmma_f32_16x16x32_bf16.
// Global->LDS staging uses gfx1250 async-copy instructions when the builtin
// is available (ASYNCcnt + s_wait_asynccnt), else global->reg->ds_store.
// ---------------------------------------------------------------------------

typedef __attribute__((ext_vector_type(16))) __bf16 bf16x16;
typedef __attribute__((ext_vector_type(8)))  __bf16 bf16x8;
typedef __attribute__((ext_vector_type(8)))  float  f32x8;
typedef int v4i __attribute__((vector_size(16)));   // builtin's pointer element type

#if defined(__HIP_DEVICE_COMPILE__) && defined(__gfx1250__) && \
    __has_builtin(__builtin_amdgcn_global_load_async_to_lds_b128)
#define USE_ASYNC_COPY 1
#else
#define USE_ASYNC_COPY 0
#endif

static __device__ __forceinline__ void wait_async0() {
#if __has_builtin(__builtin_amdgcn_s_wait_asynccnt)
    __builtin_amdgcn_s_wait_asynccnt(0);
#elif defined(__HIP_DEVICE_COMPILE__)
    asm volatile("s_wait_asynccnt 0x0" ::: "memory");
#endif
}

// ---------------------------------------------------------------------------
// Generic bf16 WMMA GEMM: C[M,N] = A[M,K] * (Bt[N,K])^T  (+bias)
// Block = 256 threads (8 waves). Block tile 128x128, k-step 32.
// Wave tile 64x32 (4x2 subtiles of 16x16). grid = (N/128, M/128, batches).
// ---------------------------------------------------------------------------
template<bool TRANS_C, bool C_IS_F32, bool BIAS>
__global__ __launch_bounds__(256)
void gemm_bf16_wmma(const __bf16* __restrict__ A,  long long strideA, int lda,
                    const __bf16* __restrict__ Bt, long long strideB, int ldb,
                    void* __restrict__ Cv,         long long strideC, int ldc,
                    const float* __restrict__ bias, int K)
{
    constexpr int LDSROW = 40;  // 32 elems + 8 pad: 80B rows -> conflict-free ds_load_b128
    __shared__ __bf16 As[2][128 * LDSROW];
    __shared__ __bf16 Bs[2][128 * LDSROW];

    const int bz = blockIdx.z;
    const __bf16* Ab = A  + (size_t)bz * strideA;
    const __bf16* Bb = Bt + (size_t)bz * strideB;
    const int row0 = blockIdx.y * 128;
    const int col0 = blockIdx.x * 128;

    const int t     = threadIdx.x;
    const int chunk = (t & 3) * 8;   // 16B chunk offset within a 32-elem row
    const int crow  = t >> 2;        // 0..63 (this thread also covers crow+64)

    const int wid  = t >> 5;
    const int wrow = (wid & 1) * 64;   // 2 wave-rows x 64 rows
    const int wcol = (wid >> 1) * 32;  // 4 wave-cols x 32 cols
    const int lane = t & 31;
    const int fr   = lane & 15;              // fragment row within 16
    const int fk   = (lane & 16) ? 8 : 0;    // fragment k base (ISA 16-bit layout)

    f32x8 acc[4][2] = {};
    const int KT = K >> 5;
    int cur = 0;

    // ---- compute one k-step from LDS buffer `buf` --------------------------
    auto compute = [&](int buf) {
        bf16x16 af[4], bfr[2];
#pragma unroll
        for (int i = 0; i < 4; ++i) {
            const __bf16* p = &As[buf][(wrow + 16 * i + fr) * LDSROW + fk];
            bf16x8 lo = *(const bf16x8*)(p);
            bf16x8 hi = *(const bf16x8*)(p + 16);
#pragma unroll
            for (int e = 0; e < 8; ++e) { af[i][e] = lo[e]; af[i][e + 8] = hi[e]; }
        }
#pragma unroll
        for (int j = 0; j < 2; ++j) {
            const __bf16* p = &Bs[buf][(wcol + 16 * j + fr) * LDSROW + fk];
            bf16x8 lo = *(const bf16x8*)(p);
            bf16x8 hi = *(const bf16x8*)(p + 16);
#pragma unroll
            for (int e = 0; e < 8; ++e) { bfr[j][e] = lo[e]; bfr[j][e + 8] = hi[e]; }
        }
#pragma unroll
        for (int i = 0; i < 4; ++i)
#pragma unroll
            for (int j = 0; j < 2; ++j)
                acc[i][j] = __builtin_amdgcn_wmma_f32_16x16x32_bf16(
                    false, af[i], false, bfr[j], (short)0, acc[i][j], false, false);
    };

#if USE_ASYNC_COPY
    // ---- async global->LDS staging (ASYNCcnt) ------------------------------
    auto stage = [&](int buf, int k0) {
        const __bf16* ga0 = Ab + (size_t)(row0 + crow)      * lda + k0 + chunk;
        const __bf16* ga1 = Ab + (size_t)(row0 + crow + 64) * lda + k0 + chunk;
        const __bf16* gb0 = Bb + (size_t)(col0 + crow)      * ldb + k0 + chunk;
        const __bf16* gb1 = Bb + (size_t)(col0 + crow + 64) * ldb + k0 + chunk;
        __bf16* la0 = &As[buf][crow * LDSROW + chunk];
        __bf16* la1 = &As[buf][(crow + 64) * LDSROW + chunk];
        __bf16* lb0 = &Bs[buf][crow * LDSROW + chunk];
        __bf16* lb1 = &Bs[buf][(crow + 64) * LDSROW + chunk];
        __builtin_amdgcn_global_load_async_to_lds_b128((v4i*)ga0, (v4i*)la0, 0, 0);
        __builtin_amdgcn_global_load_async_to_lds_b128((v4i*)ga1, (v4i*)la1, 0, 0);
        __builtin_amdgcn_global_load_async_to_lds_b128((v4i*)gb0, (v4i*)lb0, 0, 0);
        __builtin_amdgcn_global_load_async_to_lds_b128((v4i*)gb1, (v4i*)lb1, 0, 0);
    };

    stage(0, 0);
    wait_async0();
    __syncthreads();
    for (int kt = 0; kt < KT; ++kt) {
        if (kt + 1 < KT) stage(cur ^ 1, (kt + 1) * 32);   // overlap with compute
        compute(cur);
        if (kt + 1 < KT) {
            wait_async0();
            __syncthreads();
            cur ^= 1;
        }
    }
#else
    // ---- fallback: global->reg, reg->LDS (double buffered) -----------------
    bf16x8 ra0, ra1, rb0, rb1;
    auto gload = [&](int k0) {
        ra0 = *(const bf16x8*)(Ab + (size_t)(row0 + crow)      * lda + k0 + chunk);
        ra1 = *(const bf16x8*)(Ab + (size_t)(row0 + crow + 64) * lda + k0 + chunk);
        rb0 = *(const bf16x8*)(Bb + (size_t)(col0 + crow)      * ldb + k0 + chunk);
        rb1 = *(const bf16x8*)(Bb + (size_t)(col0 + crow + 64) * ldb + k0 + chunk);
    };
    auto commit = [&](int buf) {
        *(bf16x8*)&As[buf][crow * LDSROW + chunk]        = ra0;
        *(bf16x8*)&As[buf][(crow + 64) * LDSROW + chunk] = ra1;
        *(bf16x8*)&Bs[buf][crow * LDSROW + chunk]        = rb0;
        *(bf16x8*)&Bs[buf][(crow + 64) * LDSROW + chunk] = rb1;
    };
    gload(0); commit(0); __syncthreads();
    for (int kt = 0; kt < KT; ++kt) {
        if (kt + 1 < KT) gload((kt + 1) * 32);            // overlap with compute
        compute(cur);
        if (kt + 1 < KT) {
            __syncthreads();
            commit(cur ^ 1);
            __syncthreads();
            cur ^= 1;
        }
    }
#endif

    // ---- epilogue: C fragment layout: lane<16 -> col=lane, rows 0..7;
    //                lane>=16 -> col=lane-16, rows 8..15. ---------------------
    const int cn = lane & 15;
    const int rb = (lane & 16) ? 8 : 0;
#pragma unroll
    for (int i = 0; i < 4; ++i) {
#pragma unroll
        for (int j = 0; j < 2; ++j) {
            const int r0 = row0 + wrow + 16 * i + rb;
            const int c0 = col0 + wcol + 16 * j + cn;
            if (C_IS_F32) {
                float* C = (float*)Cv + (size_t)bz * strideC;
                const float bv = BIAS ? bias[c0] : 0.0f;
#pragma unroll
                for (int r = 0; r < 8; ++r)
                    C[(size_t)(r0 + r) * ldc + c0] = acc[i][j][r] + bv;
            } else if (TRANS_C) {
                // Ct[c0][r0 .. r0+7]: 8 contiguous bf16 = one 16B store/lane.
                __bf16* C = (__bf16*)Cv + (size_t)bz * strideC;
                bf16x8 v;
#pragma unroll
                for (int r = 0; r < 8; ++r) v[r] = (__bf16)acc[i][j][r];
                *(bf16x8*)(C + (size_t)c0 * ldc + r0) = v;
            } else {
                __bf16* C = (__bf16*)Cv + (size_t)bz * strideC;
#pragma unroll
                for (int r = 0; r < 8; ++r)
                    C[(size_t)(r0 + r) * ldc + c0] = (__bf16)acc[i][j][r];
            }
        }
    }
}

__global__ void cvt_f32_bf16(const float* __restrict__ in,
                             __bf16* __restrict__ out, long long n)
{
    long long i = (long long)blockIdx.x * blockDim.x + threadIdx.x;
    const long long stride = (long long)gridDim.x * blockDim.x;
    for (; i < n; i += stride) out[i] = (__bf16)in[i];
}

// Wt[c][r] = (bf16)W[r][c]   (W is rows x cols, Wt is cols x rows)
__global__ void transpose_f32_bf16(const float* __restrict__ W,
                                   __bf16* __restrict__ Wt, int rows, int cols)
{
    const int c = blockIdx.x * 16 + threadIdx.x;
    const int r = blockIdx.y * 16 + threadIdx.y;
    if (r < rows && c < cols)
        Wt[(size_t)c * rows + r] = (__bf16)W[(size_t)r * cols + c];
}

// One block per row; f32 max/sum reductions in LDS; emit bf16 probabilities.
__global__ __launch_bounds__(256)
void softmax_rows(const float* __restrict__ S, __bf16* __restrict__ P, int n)
{
    const float* row  = S + (size_t)blockIdx.x * n;
    __bf16*      prow = P + (size_t)blockIdx.x * n;
    __shared__ float red[256];

    float m = -3.402823466e38f;
    for (int c = threadIdx.x; c < n; c += 256) m = fmaxf(m, row[c]);
    red[threadIdx.x] = m; __syncthreads();
    for (int s = 128; s > 0; s >>= 1) {
        if (threadIdx.x < s)
            red[threadIdx.x] = fmaxf(red[threadIdx.x], red[threadIdx.x + s]);
        __syncthreads();
    }
    m = red[0]; __syncthreads();

    float sum = 0.0f;
    for (int c = threadIdx.x; c < n; c += 256) sum += __expf(row[c] - m);
    red[threadIdx.x] = sum; __syncthreads();
    for (int s = 128; s > 0; s >>= 1) {
        if (threadIdx.x < s) red[threadIdx.x] += red[threadIdx.x + s];
        __syncthreads();
    }
    const float inv = 1.0f / red[0];

    for (int c = threadIdx.x; c < n; c += 256)
        prow[c] = (__bf16)(__expf(row[c] - m) * inv);
}

extern "C" void kernel_launch(void* const* d_in, const int* in_sizes, int n_in,
                              void* d_out, int out_size, void* d_ws, size_t ws_size,
                              hipStream_t stream)
{
    constexpr int B = 8, N = 2048, D = 512, E = 1024;  // batch, seq, model, inner
    const float* S1 = (const float*)d_in[0];  // [B,N,D]
    const float* S2 = (const float*)d_in[1];  // [B,N,D]
    const float* Wq = (const float*)d_in[2];  // [D,E]
    const float* Wk = (const float*)d_in[3];  // [D,E]
    const float* Wv = (const float*)d_in[4];  // [D,E]
    const float* Wo = (const float*)d_in[5];  // [E,D]
    const float* bo = (const float*)d_in[6];  // [D]
    float* out = (float*)d_out;               // [B,N,D]
    (void)in_sizes; (void)n_in; (void)out_size; (void)ws_size;

    // Workspace carve-out (~356 MB), 256B aligned slices.
    char* p = (char*)d_ws;
    auto take = [&](size_t bytes) {
        char* r = p; p += (bytes + 255) & ~(size_t)255; return r;
    };
    __bf16* S1b = (__bf16*)take((size_t)B * N * D * 2);
    __bf16* S2b = (__bf16*)take((size_t)B * N * D * 2);
    __bf16* Wqt = (__bf16*)take((size_t)D * E * 2);  // [E][D]
    __bf16* Wkt = (__bf16*)take((size_t)D * E * 2);  // [E][D]
    __bf16* Wvt = (__bf16*)take((size_t)D * E * 2);  // [E][D]
    __bf16* Wot = (__bf16*)take((size_t)E * D * 2);  // [D][E]
    __bf16* Qb  = (__bf16*)take((size_t)B * N * E * 2);   // [B][N][E]
    __bf16* Kb  = (__bf16*)take((size_t)B * N * E * 2);   // [B][N][E]
    __bf16* Vt  = (__bf16*)take((size_t)B * E * N * 2);   // [B][E][N] (transposed)
    float*  Sc  = (float*) take((size_t)B * N * N * 4);   // scores f32
    __bf16* Pb  = (__bf16*)take((size_t)B * N * N * 2);   // probabilities
    __bf16* Mb  = (__bf16*)take((size_t)B * N * E * 2);   // attn output

    // 1) dtype conversions / weight transposes
    cvt_f32_bf16<<<2048, 256, 0, stream>>>(S1, S1b, (long long)B * N * D);
    cvt_f32_bf16<<<2048, 256, 0, stream>>>(S2, S2b, (long long)B * N * D);
    {
        dim3 blk(16, 16);
        transpose_f32_bf16<<<dim3(E / 16, D / 16), blk, 0, stream>>>(Wq, Wqt, D, E);
        transpose_f32_bf16<<<dim3(E / 16, D / 16), blk, 0, stream>>>(Wk, Wkt, D, E);
        transpose_f32_bf16<<<dim3(E / 16, D / 16), blk, 0, stream>>>(Wv, Wvt, D, E);
        transpose_f32_bf16<<<dim3(D / 16, E / 16), blk, 0, stream>>>(Wo, Wot, E, D);
    }

    // 2) projections: Q = S2*Wq, K = S1*Wk  -> [B][N][E] bf16
    gemm_bf16_wmma<false, false, false><<<dim3(E / 128, N / 128, B), 256, 0, stream>>>(
        S2b, (long long)N * D, D, Wqt, 0, D, Qb, (long long)N * E, E, nullptr, D);
    gemm_bf16_wmma<false, false, false><<<dim3(E / 128, N / 128, B), 256, 0, stream>>>(
        S1b, (long long)N * D, D, Wkt, 0, D, Kb, (long long)N * E, E, nullptr, D);
    //    V = S1*Wv stored transposed -> Vt [B][E][N] (so P*V's B operand is K-contiguous)
    gemm_bf16_wmma<true, false, false><<<dim3(E / 128, N / 128, B), 256, 0, stream>>>(
        S1b, (long long)N * D, D, Wvt, 0, D, Vt, (long long)E * N, N, nullptr, D);

    // 3) scores S = Q * K^T  -> f32 [B][N][N]
    gemm_bf16_wmma<false, true, false><<<dim3(N / 128, N / 128, B), 256, 0, stream>>>(
        Qb, (long long)N * E, E, Kb, (long long)N * E, E,
        Sc, (long long)N * N, N, nullptr, E);

    // 4) row softmax -> P bf16
    softmax_rows<<<B * N, 256, 0, stream>>>(Sc, Pb, N);

    // 5) M = P * V  -> bf16 [B][N][E]   (B operand = Vt rows, K-contiguous)
    gemm_bf16_wmma<false, false, false><<<dim3(E / 128, N / 128, B), 256, 0, stream>>>(
        Pb, (long long)N * N, N, Vt, (long long)E * N, N,
        Mb, (long long)N * E, E, nullptr, N);

    // 6) out = M * Wo + bo -> f32 [B][N][D]
    gemm_bf16_wmma<false, true, true><<<dim3(D / 128, N / 128, B), 256, 0, stream>>>(
        Mb, (long long)N * E, E, Wot, 0, E,
        out, (long long)N * D, D, bo, E);
}
